// CMP_32427003085021
// MI455X (gfx1250) — matrix operand order repaired and checked by reading, not compiled, not run on hardware
//
#include <hip/hip_runtime.h>

// ---------------------------------------------------------------------------
// CDNA5 (gfx1250, wave32) implementation of the graph-conv reference.
//   K0: zero pooled accumulator
//   K1: edge scatter-add (fp32 global atomics)
//   K2: build combined bf16 tensor [node][pos][64ch] (ch48-63 = 0 pad)
//   K3: spectral-norm sigma (1 power iter) + pack scaled bf16 WMMA A-fragments
//   K4: conv1 48->32 via v_wmma_f32_16x16x32_bf16, async-to-LDS staging
//   K5: conv2 32->16 via WMMA, bias+lrelu, fused residual -> out fp32
// ---------------------------------------------------------------------------

typedef __attribute__((ext_vector_type(16))) __bf16 v16bf;
typedef __attribute__((ext_vector_type(8)))  float  v8f;

#define N_NODES 2000
#define N_EDGES 8000
#define HW      1024   // 32*32

// fp32 -> bf16 round-to-nearest-even (bit pattern)
__device__ __forceinline__ unsigned short f2bf(float f) {
    union { float f; unsigned u; } x; x.f = f;
    unsigned r = x.u + 0x7FFFu + ((x.u >> 16) & 1u);
    return (unsigned short)(r >> 16);
}

// generic (flat) pointer -> LDS byte offset: LDS aperture keeps offset in addr[31:0]
__device__ __forceinline__ unsigned lds_off(const void* p) {
    return (unsigned)(unsigned long long)p;
}

// ---------------------------------------------------------------- K0: zero
__global__ void k_zero(float* __restrict__ p, int n) {
    int i = blockIdx.x * blockDim.x + threadIdx.x;
    int stride = gridDim.x * blockDim.x;
    for (; i < n; i += stride) p[i] = 0.0f;
}

// ---------------------------------------------------------------- K1: scatter
// pooled layout: [node][32ch][1024]  (ch 0-15 = pos-pool, 16-31 = neg-pool)
__global__ void k_scatter(const float* __restrict__ feats,
                          const int*   __restrict__ edges,
                          float*       __restrict__ pooled) {
    int e = blockIdx.x;
    int src = edges[e * 3 + 0];
    int typ = edges[e * 3 + 1];
    int tgt = edges[e * 3 + 2];
    if (typ == 0) return;
    src = min(max(src, 0), N_NODES - 1);
    tgt = min(max(tgt, 0), N_NODES - 1);
    int sel = (typ > 0) ? 0 : 16;                 // channel-group offset
    const float* s = feats  + (long long)src * (16 * HW);
    float*       d = pooled + (long long)tgt * (32 * HW) + (long long)sel * HW;
    for (int i = threadIdx.x; i < 16 * HW; i += blockDim.x)
        atomicAdd(&d[i], s[i]);
}

// ---------------------------------------------------------------- K2: combined
// comb layout: [node][1024 pos][64 ch] bf16;  ch0-15=feats, 16-47=pooled, 48-63=0
__global__ void k_combined(const float* __restrict__ feats,
                           const float* __restrict__ pooled,
                           unsigned short* __restrict__ comb) {
    __shared__ unsigned short lb[64][72];         // [pos][ch], padded rows
    int b       = blockIdx.x;
    int node    = b >> 4;
    int posbase = (b & 15) << 6;                  // 64 positions per block
    int tid     = threadIdx.x;

    for (int r = tid; r < 48 * 64; r += 256) {    // coalesced source reads
        int ch = r >> 6, p = r & 63;
        float v = (ch < 16)
            ? feats [((long long)node * 16 + ch)        * HW + posbase + p]
            : pooled[((long long)node * 32 + (ch - 16)) * HW + posbase + p];
        lb[p][ch] = f2bf(v);
    }
    for (int r = tid; r < 16 * 64; r += 256) {    // zero pad ch 48-63
        int ch = 48 + (r >> 6), p = r & 63;
        lb[p][ch] = 0;
    }
    __syncthreads();

    uint4* dst = (uint4*)(comb + ((long long)node * HW + posbase) * 64);
    for (int r = tid; r < 512; r += 256) {        // coalesced 16B stores
        int p = r >> 3, part = r & 7;
        dst[p * 8 + part] = *(const uint4*)&lb[p][part * 8];
    }
}

// ---------------------------------------------------------------- K3: sigma+pack
// A-fragment bf16 element layout per lane l, slot j (contiguous k order used here, B reads match):
//   K(l,j) = j + (l>=16 ? 16 : 0)
__global__ void k_sigma_pack(const float* __restrict__ W1, const float* __restrict__ u1,
                             const float* __restrict__ W2, const float* __restrict__ u2,
                             unsigned short* __restrict__ A1,
                             unsigned short* __restrict__ A2) {
    __shared__ float vbuf[432];
    __shared__ float red[256];
    __shared__ float s_inv;
    int tid = threadIdx.x;

    // ======== W1: O=32, K=432 ========
    {
        float sq = 0.f;
        for (int j = tid; j < 432; j += 256) {
            float s = 0.f;
            for (int o = 0; o < 32; ++o) s += W1[o * 432 + j] * u1[o];
            vbuf[j] = s; sq += s * s;
        }
        red[tid] = sq; __syncthreads();
        for (int k = 128; k > 0; k >>= 1) {
            if (tid < k) red[tid] += red[tid + k];
            __syncthreads();
        }
        float inv = 1.f / (sqrtf(red[0]) + 1e-12f);
        __syncthreads();
        if (tid < 32) {
            float wv = 0.f;
            for (int j = 0; j < 432; ++j) wv += W1[tid * 432 + j] * (vbuf[j] * inv);
            red[tid] = wv * wv;
        }
        __syncthreads();
        if (tid == 0) {
            float ss = 0.f;
            for (int o = 0; o < 32; ++o) ss += red[o];
            float n2 = sqrtf(ss);
            float sigma = ss / (n2 + 1e-12f);
            s_inv = 1.f / sigma;
        }
        __syncthreads();
        float invs = s_inv;
        // pack A1: linear = ((mt*9 + t)*2 + c)*512 + l*16 + j   (18432 total)
        for (int i = tid; i < 18432; i += 256) {
            int j  = i & 15;
            int l  = (i >> 4) & 31;
            int c  = (i >> 9) & 1;
            int t  = (i >> 10) % 9;
            int mt = (i >> 10) / 9;
            int M  = mt * 16 + (l & 15);
            int K  = c * 32 + j + ((l >= 16) ? 16 : 0);
            float w = 0.f;
            if (K < 48) {
                int kh = t / 3, kw = t % 3;
                w = W1[((M * 48 + K) * 3 + kh) * 3 + kw] * invs;
            }
            A1[i] = f2bf(w);
        }
    }
    __syncthreads();
    // ======== W2: O=16, K=288 ========
    {
        float sq = 0.f;
        for (int j = tid; j < 288; j += 256) {
            float s = 0.f;
            for (int o = 0; o < 16; ++o) s += W2[o * 288 + j] * u2[o];
            vbuf[j] = s; sq += s * s;
        }
        red[tid] = sq; __syncthreads();
        for (int k = 128; k > 0; k >>= 1) {
            if (tid < k) red[tid] += red[tid + k];
            __syncthreads();
        }
        float inv = 1.f / (sqrtf(red[0]) + 1e-12f);
        __syncthreads();
        if (tid < 16) {
            float wv = 0.f;
            for (int j = 0; j < 288; ++j) wv += W2[tid * 288 + j] * (vbuf[j] * inv);
            red[tid] = wv * wv;
        }
        __syncthreads();
        if (tid == 0) {
            float ss = 0.f;
            for (int o = 0; o < 16; ++o) ss += red[o];
            float n2 = sqrtf(ss);
            float sigma = ss / (n2 + 1e-12f);
            s_inv = 1.f / sigma;
        }
        __syncthreads();
        float invs = s_inv;
        // pack A2: linear = t*512 + l*16 + j   (4608 total)
        for (int i = tid; i < 4608; i += 256) {
            int j = i & 15;
            int l = (i >> 4) & 31;
            int t = i >> 9;
            int M = (l & 15);
            int K = j + ((l >= 16) ? 16 : 0);
            int kh = t / 3, kw = t % 3;
            A2[i] = f2bf(W2[((M * 32 + K) * 3 + kh) * 3 + kw] * invs);
        }
    }
}

// ---------------------------------------------------------------- K4: conv1
// comb [node][pos][64] bf16 -> h1 [node][pos][32] bf16 (bias + lrelu applied)
__global__ void __launch_bounds__(256)
k_conv1(const unsigned short* __restrict__ comb,
        const unsigned short* __restrict__ A1,
        const float*          __restrict__ b1,
        unsigned short*       __restrict__ h1) {
    __shared__ __align__(32) unsigned short tile[10 * 34 * 64]; // [row][col][ch]
    int node = blockIdx.x;
    int r0   = blockIdx.y * 8;                    // 8 output rows per block
    int tid  = threadIdx.x;

    // stage with halo: async global->LDS for valid texels, zero-store the border
    for (int i = tid; i < 2720; i += 256) {       // 10 rows x 34 cols x 128B
        int part = i & 7;
        int seg  = i >> 3;
        int lrow = seg / 34;
        int lcol = seg - lrow * 34;
        int grow = r0 - 1 + lrow;
        int gcol = lcol - 1;
        unsigned short* dst = &tile[(lrow * 34 + lcol) * 64 + part * 8];
        if (grow >= 0 && grow < 32 && gcol >= 0 && gcol < 32) {
            unsigned goff = (unsigned)((((node * HW + grow * 32 + gcol) << 6)
                                        + part * 8) * 2);
            asm volatile("global_load_async_to_lds_b128 %0, %1, %2"
                         :: "v"(lds_off(dst)), "v"(goff), "s"(comb) : "memory");
        } else {
            *((uint4*)dst) = make_uint4(0u, 0u, 0u, 0u);
        }
    }
    asm volatile("s_wait_asynccnt 0x0" ::: "memory");
    __syncthreads();

    int wave  = tid >> 5;
    int lane  = tid & 31;
    int mt    = wave & 1;                         // M tile (o 0-15 / 16-31)
    int ng    = wave >> 1;                        // N-tile group: nt = ng*4..+3
    int lhalf = lane >> 4;
    int ln    = lane & 15;

    v8f zero = {0.f, 0.f, 0.f, 0.f, 0.f, 0.f, 0.f, 0.f};
    v8f acc[4] = {zero, zero, zero, zero};

    const v16bf* Abase = (const v16bf*)A1;
    for (int t = 0; t < 9; ++t) {
        int kh = t / 3, kw = t % 3;
#pragma unroll
        for (int c = 0; c < 2; ++c) {
            v16bf a = Abase[((mt * 9 + t) * 2 + c) * 32 + lane];
            int chbase = c * 32 + lhalf * 16;
            v16bf bfr[4];
#pragma unroll
            for (int k = 0; k < 4; ++k) {         // batch B loads first
                int nt   = ng * 4 + k;
                int lrow = (nt >> 1) + kh;
                int lcol = ((nt & 1) << 4) + ln + kw;
                bfr[k] = *(const v16bf*)&tile[(lrow * 34 + lcol) * 64 + chbase];
            }
#pragma unroll
            for (int k = 0; k < 4; ++k)
                acc[k] = __builtin_amdgcn_wmma_f32_16x16x32_bf16(
                    false, a, false, bfr[k], (short)0, acc[k], false, false);
        }
    }

    for (int k = 0; k < 4; ++k) {
        int nt   = ng * 4 + k;
        int prow = r0 + (nt >> 1);
        int pcol = ((nt & 1) << 4) + ln;
        unsigned short* dst = h1 + ((long long)node * HW + prow * 32 + pcol) * 32;
        for (int g = 0; g < 8; g += 2) {
            int o = mt * 16 + g + lhalf * 8;
            float v0 = acc[k][g]     + b1[o];
            float v1 = acc[k][g + 1] + b1[o + 1];
            v0 = v0 > 0.f ? v0 : 0.2f * v0;
            v1 = v1 > 0.f ? v1 : 0.2f * v1;
            unsigned pk = (unsigned)f2bf(v0) | ((unsigned)f2bf(v1) << 16);
            *((unsigned*)&dst[o]) = pk;
        }
    }
}

// ---------------------------------------------------------------- K5: conv2 (+residual)
__global__ void __launch_bounds__(256)
k_conv2(const unsigned short* __restrict__ h1,
        const unsigned short* __restrict__ A2,
        const float*          __restrict__ b2,
        const float*          __restrict__ feats,
        float*                __restrict__ out) {
    __shared__ __align__(32) unsigned short tile[10 * 34 * 32]; // [row][col][ch]
    int node = blockIdx.x;
    int r0   = blockIdx.y * 8;
    int tid  = threadIdx.x;

    for (int i = tid; i < 1360; i += 256) {       // 10*34 segs x 64B
        int part = i & 3;
        int seg  = i >> 2;
        int lrow = seg / 34;
        int lcol = seg - lrow * 34;
        int grow = r0 - 1 + lrow;
        int gcol = lcol - 1;
        unsigned short* dst = &tile[(lrow * 34 + lcol) * 32 + part * 8];
        if (grow >= 0 && grow < 32 && gcol >= 0 && gcol < 32) {
            unsigned goff = (unsigned)((((node * HW + grow * 32 + gcol) << 5)
                                        + part * 8) * 2);
            asm volatile("global_load_async_to_lds_b128 %0, %1, %2"
                         :: "v"(lds_off(dst)), "v"(goff), "s"(h1) : "memory");
        } else {
            *((uint4*)dst) = make_uint4(0u, 0u, 0u, 0u);
        }
    }
    asm volatile("s_wait_asynccnt 0x0" ::: "memory");
    __syncthreads();

    int wave  = tid >> 5;
    int lane  = tid & 31;
    int lhalf = lane >> 4;
    int ln    = lane & 15;

    v8f zero = {0.f, 0.f, 0.f, 0.f, 0.f, 0.f, 0.f, 0.f};
    v8f acc[2] = {zero, zero};

    const v16bf* Abase = (const v16bf*)A2;
    v16bf afr[9];
#pragma unroll
    for (int t = 0; t < 9; ++t) afr[t] = Abase[t * 32 + lane];  // preload all A

    int chbase = lhalf * 16;
#pragma unroll
    for (int t = 0; t < 9; ++t) {
        int kh = t / 3, kw = t % 3;
        v16bf bfr[2];
#pragma unroll
        for (int k = 0; k < 2; ++k) {             // batch B loads first
            int nt   = wave * 2 + k;
            int lrow = (nt >> 1) + kh;
            int lcol = ((nt & 1) << 4) + ln + kw;
            bfr[k] = *(const v16bf*)&tile[(lrow * 34 + lcol) * 32 + chbase];
        }
#pragma unroll
        for (int k = 0; k < 2; ++k)
            acc[k] = __builtin_amdgcn_wmma_f32_16x16x32_bf16(
                false, afr[t], false, bfr[k], (short)0, acc[k], false, false);
    }

    for (int k = 0; k < 2; ++k) {
        int nt   = wave * 2 + k;
        int pos  = (r0 + (nt >> 1)) * 32 + ((nt & 1) << 4) + ln;
        for (int g = 0; g < 8; ++g) {
            int o = g + lhalf * 8;
            float v = acc[k][g] + b2[o];
            v = v > 0.f ? v : 0.2f * v;
            long long idx = ((long long)node * 16 + o) * HW + pos;
            out[idx] = feats[idx] + v;
        }
    }
}

// ---------------------------------------------------------------- launch
extern "C" void kernel_launch(void* const* d_in, const int* in_sizes, int n_in,
                              void* d_out, int out_size, void* d_ws, size_t ws_size,
                              hipStream_t stream) {
    const float* feats = (const float*)d_in[0];
    const int*   edges = (const int*)  d_in[1];
    const float* W1    = (const float*)d_in[2];
    const float* b1    = (const float*)d_in[3];
    const float* u1    = (const float*)d_in[4];
    const float* W2    = (const float*)d_in[5];
    const float* b2    = (const float*)d_in[6];
    const float* u2    = (const float*)d_in[7];
    float* out = (float*)d_out;

    char* ws = (char*)d_ws;
    const long long POOLED_BYTES = (long long)N_NODES * 32 * HW * 4; // 262,144,000
    const long long COMB_BYTES   = (long long)N_NODES * HW * 64 * 2; // 262,144,000
    float*          pooled = (float*)ws;
    unsigned short* h1     = (unsigned short*)ws;            // aliases pooled (dead by then)
    unsigned short* comb   = (unsigned short*)(ws + POOLED_BYTES);
    unsigned short* A1     = (unsigned short*)(ws + POOLED_BYTES + COMB_BYTES);
    unsigned short* A2     = (unsigned short*)(ws + POOLED_BYTES + COMB_BYTES + 65536);

    int pooled_elems = N_NODES * 32 * HW;                    // 65,536,000
    k_zero<<<8192, 256, 0, stream>>>(pooled, pooled_elems);
    k_scatter<<<N_EDGES, 256, 0, stream>>>(feats, edges, pooled);
    k_combined<<<N_NODES * 16, 256, 0, stream>>>(feats, pooled, comb);
    k_sigma_pack<<<1, 256, 0, stream>>>(W1, u1, W2, u2, A1, A2);
    k_conv1<<<dim3(N_NODES, 4), 256, 0, stream>>>(comb, A1, b1, h1);
    k_conv2<<<dim3(N_NODES, 4), 256, 0, stream>>>(h1, A2, b2, feats, out);
}
